// GraphformerEncoderLayer_15479062135328
// MI455X (gfx1250) — matrix operand order repaired and checked by reading, not compiled, not run on hardware
//
#include <hip/hip_runtime.h>
#include <math.h>
#include <stdint.h>

// ---------- types ----------
typedef __bf16 bf16_t;
typedef bf16_t v8bf  __attribute__((ext_vector_type(8)));
typedef bf16_t v16bf __attribute__((ext_vector_type(16)));
typedef float  v8f   __attribute__((ext_vector_type(8)));
typedef unsigned int u32x4 __attribute__((ext_vector_type(4)));
typedef int i32x4 __attribute__((ext_vector_type(4)));
typedef int i32x8 __attribute__((ext_vector_type(8)));

#define WMMA_BF16(a, b, c) \
  __builtin_amdgcn_wmma_f32_16x16x32_bf16(false, (a), false, (b), (short)0, (c), false, false)

static __device__ __forceinline__ v16bf cat16(v8bf lo, v8bf hi) {
  return __builtin_shufflevector(lo, hi, 0,1,2,3,4,5,6,7,8,9,10,11,12,13,14,15);
}
// A-fragment (16x32 bf16): lane half h, row r: K = h*8+0..7 and 16+h*8+0..7  (ISA 7.12.2)
static __device__ __forceinline__ v16bf load_a_frag(const bf16_t* row_k0, int half) {
  v8bf lo = *(const v8bf*)(row_k0 + half * 8);
  v8bf hi = *(const v8bf*)(row_k0 + 16 + half * 8);
  return cat16(lo, hi);
}
// B-fragment (32x16 bf16): lane half h, column n: K = h*8+0..7 and 16+h*8+0..7 (strided)
static __device__ __forceinline__ v16bf load_b_frag(const bf16_t* p, int stride, int half) {
  v16bf v;
#pragma unroll
  for (int s = 0; s < 8; s++) {
    v[s] = p[(size_t)(half * 8 + s) * stride];
    v[8 + s] = p[(size_t)(16 + half * 8 + s) * stride];
  }
  return v;
}

// ---------- TDM: 2D tensor tile -> LDS (D# per ISA 8.3/8.4), bf16 elements ----------
// pad_interval_code: pad after (2<<code) DWORDs; pad_amount_code: insert (code+1) DWORDs
static __device__ __forceinline__ void tdm_load_2d(unsigned lds_off, const void* gaddr,
                                                   unsigned tile_d0, unsigned tile_d1,
                                                   unsigned tensor_d0, unsigned tensor_d1,
                                                   unsigned stride0_elems,
                                                   unsigned pad_interval_code,
                                                   unsigned pad_amount_code) {
  unsigned long long ga = (unsigned long long)(uintptr_t)gaddr;
  u32x4 g0;
  g0[0] = 1u;                                            // count=1 (valid user D#)
  g0[1] = lds_off;                                       // LDS byte address
  g0[2] = (unsigned)(ga & 0xffffffffu);                  // global_addr[31:0]
  g0[3] = (unsigned)((ga >> 32) & 0x01ffffffu) | (2u << 30); // global_addr[56:32] | type=2
  i32x8 g1;
  g1[0] = (int)((1u << 16) | (1u << 20) |                // data_size=2B, pad_enable
                (pad_interval_code << 22) | (pad_amount_code << 25));
  g1[1] = (int)((tensor_d0 & 0xffffu) << 16);            // tensor_dim0[15:0] (atomic addr=0)
  g1[2] = (int)(((tensor_d0 >> 16) & 0xffffu) | ((tensor_d1 & 0xffffu) << 16));
  g1[3] = (int)(((tensor_d1 >> 16) & 0xffffu) | ((tile_d0 & 0xffffu) << 16));
  g1[4] = (int)(tile_d1 & 0xffffu);                      // tile_dim1 | tile_dim2=0
  g1[5] = (int)stride0_elems;                            // tensor_dim0_stride[31:0]
  g1[6] = 0;
  g1[7] = 0;
  i32x4 gz = {0, 0, 0, 0};
#if defined(__clang_major__) && (__clang_major__ >= 23)
  i32x8 gz8 = {0, 0, 0, 0, 0, 0, 0, 0};
  __builtin_amdgcn_tensor_load_to_lds(g0, g1, gz, gz, gz8, 0);
#else
  __builtin_amdgcn_tensor_load_to_lds(g0, g1, gz, gz, 0);
#endif
}

// ---------- weight cast + transpose: in[R][C] f32 -> out[C][R] bf16 ----------
__global__ __launch_bounds__(256) void k_transpose_cast(const float* __restrict__ in,
                                                        bf16_t* __restrict__ out,
                                                        int R, int C) {
  int idx = blockIdx.x * 256 + threadIdx.x;      // linear over output (coalesced writes)
  if (idx >= R * C) return;
  int r = idx % R, c = idx / R;
  out[idx] = (bf16_t)in[(size_t)r * C + c];
}

// ---------- bf16 transpose (tiled): in[R][C] -> out[C][R] ----------
__global__ __launch_bounds__(256) void k_transpose_bf16(const bf16_t* __restrict__ in,
                                                        bf16_t* __restrict__ out,
                                                        int R, int C) {
  __shared__ bf16_t tile[32][33];
  int c0 = blockIdx.x * 32, r0 = blockIdx.y * 32;
  int x = threadIdx.x, y = threadIdx.y;          // block (32,8)
#pragma unroll
  for (int i = 0; i < 32; i += 8)
    tile[y + i][x] = in[(size_t)(r0 + y + i) * C + c0 + x];
  __syncthreads();
#pragma unroll
  for (int i = 0; i < 32; i += 8)
    out[(size_t)(c0 + y + i) * R + r0 + x] = tile[x][y + i];
}

// ---------- LayerNorm + cast to bf16 (one wave per row, D=256) ----------
__global__ __launch_bounds__(256) void k_ln_cast(const float* __restrict__ x,
                                                 const float* __restrict__ gam,
                                                 const float* __restrict__ bet,
                                                 bf16_t* __restrict__ out, int D) {
  int wave = threadIdx.x >> 5, lane = threadIdx.x & 31;
  int row = blockIdx.x * 8 + wave;
  const float* xr = x + (size_t)row * D;
  float v[8], s = 0.f, s2 = 0.f;
#pragma unroll
  for (int i = 0; i < 8; i++) {
    v[i] = xr[i * 32 + lane];
    s += v[i]; s2 += v[i] * v[i];
  }
#pragma unroll
  for (int m = 1; m < 32; m <<= 1) {
    s  += __shfl_xor(s,  m, 32);
    s2 += __shfl_xor(s2, m, 32);
  }
  float mean = s / (float)D;
  float var  = s2 / (float)D - mean * mean;
  float rstd = rsqrtf(var + 1e-5f);
#pragma unroll
  for (int i = 0; i < 8; i++) {
    int c = i * 32 + lane;
    out[(size_t)row * D + c] = (bf16_t)((v[i] - mean) * rstd * gam[c] + bet[c]);
  }
}

// ---------- generic WMMA GEMM: out[M][Nn] = A[M][K](bf16) @ Bt[K][Nn](bf16) ----------
// EPI 0: out bf16 = acc
// EPI 1: out f32  = acc + bias[col] + res[off]
// EPI 2: out bf16 = gelu_exact(acc + bias[col])
template <int EPI>
__global__ __launch_bounds__(256) void k_gemm_bt(const bf16_t* __restrict__ A,
                                                 const bf16_t* __restrict__ Bt,
                                                 const float* __restrict__ bias,
                                                 const float* __restrict__ res,
                                                 void* __restrict__ outv,
                                                 int M, int K, int Nn) {
  int wave = threadIdx.x >> 5, lane = threadIdx.x & 31;
  int half = lane >> 4, r = lane & 15;
  int m0 = blockIdx.x * 128 + wave * 16;
  int n0 = blockIdx.y * 64;
  v8f acc[4] = {};
  const bf16_t* arow = A + (size_t)(m0 + r) * K;
  for (int k0 = 0; k0 < K; k0 += 32) {
    v16bf af = load_a_frag(arow + k0, half);
    const bf16_t* brow = Bt + (size_t)k0 * Nn + n0 + r;
#pragma unroll
    for (int t = 0; t < 4; t++) {
      v16bf bfrg = load_b_frag(brow + t * 16, Nn, half);
      acc[t] = WMMA_BF16(af, bfrg, acc[t]);
    }
  }
  int rowbase = m0 + half * 8;
#pragma unroll
  for (int t = 0; t < 4; t++) {
    int col = n0 + t * 16 + r;
#pragma unroll
    for (int g = 0; g < 8; g++) {
      size_t off = (size_t)(rowbase + g) * Nn + col;
      float v = acc[t][g];
      if (EPI == 0) {
        ((bf16_t*)outv)[off] = (bf16_t)v;
      } else if (EPI == 1) {
        ((float*)outv)[off] = v + bias[col] + res[off];
      } else {
        v += bias[col];
        ((bf16_t*)outv)[off] = (bf16_t)(0.5f * v * (1.0f + erff(v * 0.70710678118654752f)));
      }
    }
  }
}

// ---------- flash attention (per head, d_head=256, N=4096) ----------
// Qb/Vb/Ob: [N][2048] bf16 (head h at col h*256). Kt: [2048][N] bf16 (transposed K).
// K^T/V 64-key tiles are TDM-DMA'd into double-buffered LDS (shared by all 8 waves),
// overlapping the next tile's DMA with WMMA on the current tile.
__global__ __launch_bounds__(256) void k_flash_attn(const bf16_t* __restrict__ Qb,
                                                    const bf16_t* __restrict__ Kt,
                                                    const bf16_t* __restrict__ Vb,
                                                    bf16_t* __restrict__ Ob,
                                                    int NKV) {
  const int HD = 2048;
  // TDM pads each 64-elem K row by 8 elems (->72) and each 256-elem V row by 8 (->264)
  __shared__ __align__(16) bf16_t tileK[2][256][72];
  __shared__ __align__(16) bf16_t tileV[2][64][264];
  __shared__ __align__(16) bf16_t plds[8][16][72];
  int wave = threadIdx.x >> 5, lane = threadIdx.x & 31;
  int half = lane >> 4, r = lane & 15;
  int h = blockIdx.y;
  int m0 = blockIdx.x * 128 + wave * 16;

  // Q fragments resident in VGPRs: 16 rows x 256 features = 8 chunks
  v16bf qf[8];
  const bf16_t* qrow = Qb + (size_t)(m0 + r) * HD + h * 256;
#pragma unroll
  for (int kk = 0; kk < 8; kk++) qf[kk] = load_a_frag(qrow + kk * 32, half);

  v8f o[16] = {};
  float mrun[8], lrun[8];
#pragma unroll
  for (int g = 0; g < 8; g++) { mrun[g] = -1e30f; lrun[g] = 0.f; }
  const float scale = 0.0625f;   // 1/sqrt(256)

  const bf16_t* kt_head = Kt + (size_t)(h * 256) * NKV;   // [256][NKV]
  const bf16_t* v_head  = Vb + (size_t)h * 256;           // rows stride HD

  int nj = NKV / 64;
  // prologue: DMA tile 0 into buffer 0 (wave 0 only; TDM ignores EXEC, branch is uniform)
  if (threadIdx.x < 32) {
    tdm_load_2d((unsigned)(uintptr_t)&tileK[0][0][0], kt_head,
                /*tile*/ 64, 256, /*tensor*/ NKV, 256, /*stride0*/ NKV,
                /*pad: every 32 dw*/ 4, /*+4 dw*/ 3);
    tdm_load_2d((unsigned)(uintptr_t)&tileV[0][0][0], v_head,
                /*tile*/ 256, 64, /*tensor*/ HD, NKV, /*stride0*/ HD,
                /*pad: every 128 dw*/ 6, /*+4 dw*/ 3);
  }

  for (int jb = 0; jb < nj; jb++) {
    int cur = jb & 1;
    if (threadIdx.x < 32) __builtin_amdgcn_s_wait_tensorcnt(0);
    __syncthreads();   // tiles[cur] ready; everyone done reading tiles[cur^1]
    if (threadIdx.x < 32 && (jb + 1) < nj) {
      int j1 = (jb + 1) * 64;
      tdm_load_2d((unsigned)(uintptr_t)&tileK[cur ^ 1][0][0], kt_head + j1,
                  64, 256, NKV, 256, NKV, 4, 3);
      tdm_load_2d((unsigned)(uintptr_t)&tileV[cur ^ 1][0][0], v_head + (size_t)j1 * HD,
                  256, 64, HD, NKV, HD, 6, 3);
    }

    // S = Q @ K^T for 64 keys (K tile from LDS)
    v8f s[4] = {};
#pragma unroll
    for (int kk = 0; kk < 8; kk++) {
      const bf16_t* krow = &tileK[cur][kk * 32][r];
#pragma unroll
      for (int t = 0; t < 4; t++) {
        v16bf bfrg = load_b_frag(krow + t * 16, 72, half);
        s[t] = WMMA_BF16(qf[kk], bfrg, s[t]);
      }
    }
    // online softmax: rows live in 16-lane halves; reduce with xor shuffles
    float alpha[8];
#pragma unroll
    for (int g = 0; g < 8; g++) {
      float mx = -1e30f;
#pragma unroll
      for (int t = 0; t < 4; t++) { s[t][g] *= scale; mx = fmaxf(mx, s[t][g]); }
#pragma unroll
      for (int m = 1; m < 16; m <<= 1) mx = fmaxf(mx, __shfl_xor(mx, m, 32));
      float mnew = fmaxf(mrun[g], mx);
      alpha[g] = __expf(mrun[g] - mnew);
      mrun[g] = mnew;
      float rs = 0.f;
#pragma unroll
      for (int t = 0; t < 4; t++) {
        float pv = __expf(s[t][g] - mnew);
        s[t][g] = pv; rs += pv;
      }
#pragma unroll
      for (int m = 1; m < 16; m <<= 1) rs += __shfl_xor(rs, m, 32);
      lrun[g] = lrun[g] * alpha[g] + rs;
    }
    // rescale O accumulator
#pragma unroll
    for (int dt = 0; dt < 16; dt++)
#pragma unroll
      for (int g = 0; g < 8; g++) o[dt][g] *= alpha[g];
    // P: D-layout -> A-fragment via LDS (per-wave region, in-order DS)
#pragma unroll
    for (int t = 0; t < 4; t++)
#pragma unroll
      for (int g = 0; g < 8; g++)
        plds[wave][half * 8 + g][t * 16 + r] = (bf16_t)s[t][g];
    // O += P @ V (V tile from LDS)
#pragma unroll
    for (int jc = 0; jc < 2; jc++) {
      const v8bf* prow = (const v8bf*)&plds[wave][r][0];
      v16bf pf = cat16(prow[(jc * 32 + half * 8) >> 3],
                       prow[(jc * 32 + 16 + half * 8) >> 3]);
      const bf16_t* vrow = &tileV[cur][jc * 32][r];
#pragma unroll
      for (int dt = 0; dt < 16; dt++) {
        v16bf vf = load_b_frag(vrow + dt * 16, 264, half);
        o[dt] = WMMA_BF16(pf, vf, o[dt]);
      }
    }
  }
  // normalize + store
  int rowbase = m0 + half * 8;
#pragma unroll
  for (int dt = 0; dt < 16; dt++) {
    int col = h * 256 + dt * 16 + r;
#pragma unroll
    for (int g = 0; g < 8; g++)
      Ob[(size_t)(rowbase + g) * HD + col] = (bf16_t)(o[dt][g] / lrun[g]);
  }
}

// ---------- host ----------
extern "C" void kernel_launch(void* const* d_in, const int* in_sizes, int n_in,
                              void* d_out, int out_size, void* d_ws, size_t ws_size,
                              hipStream_t stream) {
  const float* x   = (const float*)d_in[0];
  const float* Wq  = (const float*)d_in[1];
  const float* Wk  = (const float*)d_in[2];
  const float* Wv  = (const float*)d_in[3];
  const float* Wo  = (const float*)d_in[4];
  const float* bo  = (const float*)d_in[5];
  const float* g1  = (const float*)d_in[6];
  const float* be1 = (const float*)d_in[7];
  const float* g2  = (const float*)d_in[8];
  const float* be2 = (const float*)d_in[9];
  const float* W1  = (const float*)d_in[10];
  const float* b1  = (const float*)d_in[11];
  const float* W2  = (const float*)d_in[12];
  const float* b2  = (const float*)d_in[13];
  float* out = (float*)d_out;

  const int N = 4096, D = 256, HD = 2048, HID = 1024;

  char* p = (char*)d_ws;
  auto alloc = [&](size_t bytes) -> void* {
    void* r = (void*)p;
    p += (bytes + 255) & ~(size_t)255;
    return r;
  };
  bf16_t* Wqt = (bf16_t*)alloc((size_t)D * HD * 2);    // [256][2048]
  bf16_t* Wkt = (bf16_t*)alloc((size_t)D * HD * 2);
  bf16_t* Wvt = (bf16_t*)alloc((size_t)D * HD * 2);
  bf16_t* Wot = (bf16_t*)alloc((size_t)HD * D * 2);    // [2048][256]
  bf16_t* W1t = (bf16_t*)alloc((size_t)D * HID * 2);   // [256][1024]
  bf16_t* W2t = (bf16_t*)alloc((size_t)HID * D * 2);   // [1024][256]
  bf16_t* h   = (bf16_t*)alloc((size_t)N * D * 2);
  bf16_t* Qb  = (bf16_t*)alloc((size_t)N * HD * 2);
  bf16_t* Kb  = (bf16_t*)alloc((size_t)N * HD * 2);
  bf16_t* Vb  = (bf16_t*)alloc((size_t)N * HD * 2);
  bf16_t* Ktr = (bf16_t*)alloc((size_t)HD * N * 2);
  bf16_t* att = (bf16_t*)alloc((size_t)N * HD * 2);
  float*  x1  = (float*)alloc((size_t)N * D * 4);
  bf16_t* h2  = (bf16_t*)alloc((size_t)N * D * 2);
  bf16_t* hid = (bf16_t*)alloc((size_t)N * HID * 2);

  // weights: cast+transpose to bf16 [K][Nn]
  k_transpose_cast<<<(HD * D) / 256, 256, 0, stream>>>(Wq, Wqt, HD, D);
  k_transpose_cast<<<(HD * D) / 256, 256, 0, stream>>>(Wk, Wkt, HD, D);
  k_transpose_cast<<<(HD * D) / 256, 256, 0, stream>>>(Wv, Wvt, HD, D);
  k_transpose_cast<<<(D * HD) / 256, 256, 0, stream>>>(Wo, Wot, D, HD);
  k_transpose_cast<<<(HID * D) / 256, 256, 0, stream>>>(W1, W1t, HID, D);
  k_transpose_cast<<<(D * HID) / 256, 256, 0, stream>>>(W2, W2t, D, HID);

  // LN1
  k_ln_cast<<<N / 8, 256, 0, stream>>>(x, g1, be1, h, D);

  // Q, K, V projections
  dim3 gqkv(N / 128, HD / 64);
  k_gemm_bt<0><<<gqkv, 256, 0, stream>>>(h, Wqt, nullptr, nullptr, Qb, N, D, HD);
  k_gemm_bt<0><<<gqkv, 256, 0, stream>>>(h, Wkt, nullptr, nullptr, Kb, N, D, HD);
  k_gemm_bt<0><<<gqkv, 256, 0, stream>>>(h, Wvt, nullptr, nullptr, Vb, N, D, HD);

  // K transpose for score WMMAs
  dim3 gtr(HD / 32, N / 32);
  k_transpose_bf16<<<gtr, dim3(32, 8), 0, stream>>>(Kb, Ktr, N, HD);

  // flash attention (TDM-staged K/V tiles)
  k_flash_attn<<<dim3(N / 128, 8), 256, 0, stream>>>(Qb, Ktr, Vb, att, N);

  // Wo projection + bias + residual -> x1 (f32)
  k_gemm_bt<1><<<dim3(N / 128, D / 64), 256, 0, stream>>>(att, Wot, bo, x, x1, N, HD, D);

  // LN2
  k_ln_cast<<<N / 8, 256, 0, stream>>>(x1, g2, be2, h2, D);

  // FFN
  k_gemm_bt<2><<<dim3(N / 128, HID / 64), 256, 0, stream>>>(h2, W1t, b1, nullptr, hid, N, D, HID);
  k_gemm_bt<1><<<dim3(N / 128, D / 64), 256, 0, stream>>>(hid, W2t, b2, x1, out, N, HID, D);
}
